// PCDAlignment_47863115546743
// MI455X (gfx1250) — matrix operand order repaired and checked by reading, not compiled, not run on hardware
//
#include <hip/hip_runtime.h>

// ---------------------------------------------------------------------------
// PCD alignment (EDVR-style) for MI455X / gfx1250.
// All intermediates are NHWC bf16 (channel-contiguous records) so:
//  - conv LDS staging is a straight 16B-chunk copy via
//    global_load_async_to_lds_b128 (ASYNCcnt path, no VALU conversion)
//  - conv epilogues store 8 consecutive couts with one global_store_b128
//  - deformable-conv bilinear corners load 8 channels with one b128 load
// All GEMMs run on v_wmma_f32_16x16x32_bf16 (wave32).
// bf16 packing uses native __bf16 converts (v_cvt_pk_bf16_f32, RNE).
// ---------------------------------------------------------------------------

#define USE_ASYNC_STAGING 1

typedef __attribute__((ext_vector_type(16))) __bf16 v16bf;
typedef __attribute__((ext_vector_type(2)))  __bf16 v2bf;
typedef __attribute__((ext_vector_type(8)))  float  v8f;
typedef unsigned short u16t;

__device__ __forceinline__ u16t f32_to_bf16(float f) {
  __bf16 b = (__bf16)f;                       // native cvt, RNE
  return __builtin_bit_cast(u16t, b);
}
__device__ __forceinline__ unsigned pack2(float lo, float hi) {
  v2bf v = { (__bf16)lo, (__bf16)hi };        // v_cvt_pk_bf16_f32
  return __builtin_bit_cast(unsigned, v);
}
__device__ __forceinline__ float bf2f(u16t u) {
  return __uint_as_float((unsigned)u << 16);
}
__device__ __forceinline__ float bfl(unsigned u) {   // low  bf16 of a dword
  return __uint_as_float(u << 16);
}
__device__ __forceinline__ float bfh(unsigned u) {   // high bf16 of a dword
  return __uint_as_float(u & 0xffff0000u);
}
__device__ __forceinline__ int iclamp(int v, int lo, int hi) {
  return v < lo ? lo : (v > hi ? hi : v);
}

// ---------------------------------------------------------------------------
// Weight pre-pack: f32 (Cout,Cin,3,3) -> bf16 A-fragments for the
// 16x32 bf16 A operand, contiguous k order (k = 16*(lane/16) + element).  One uint = one K-pair.
// Layout: frag[((chunk*MB + mblk)*256 + lane*8 + r)]
//   conv mode: K = tap*Cin + cin          (tap-major, cin within chunk)
//   dcn  mode: K = (g*9 + k9)*8 + c, cin = g*8 + c, tap = k9
// ---------------------------------------------------------------------------
__global__ void prepack_wmma(const float* __restrict__ w,
                             unsigned int* __restrict__ frag,
                             int Cin, int MB, int CoutReal, int Ktot, int dcn) {
  int idx = blockIdx.x * 256 + threadIdx.x;
  int total = (Ktot / 32) * MB * 256;
  if (idx >= total) return;
  int r     = idx & 7;
  int lane  = (idx >> 3) & 31;
  int rest  = idx >> 8;
  int m     = rest % MB;
  int chunk = rest / MB;
  int mrow  = m * 16 + (lane & 15);
  int khalf = lane >> 4;
  int kbase = khalf ? 16 : 0;
  int kloc  = kbase + 2 * r;
  float vv[2];
#pragma unroll
  for (int j = 0; j < 2; ++j) {
    int Kg = chunk * 32 + kloc + j;
    float v = 0.f;
    if (mrow < CoutReal && Kg < Ktot) {
      int cin, tap;
      if (dcn) {
        int pair = Kg >> 3, c = Kg & 7;
        int g = pair / 9, k9 = pair - g * 9;
        cin = g * 8 + c; tap = k9;
      } else {
        tap = Kg / Cin; cin = Kg - tap * Cin;
      }
      v = w[((long)mrow * Cin + cin) * 9 + tap];
    }
    vv[j] = v;
  }
  frag[idx] = pack2(vv[0], vv[1]);
}

// ---------------------------------------------------------------------------
// One M-pass (NMB x 16 output channels) of the implicit-GEMM conv.
// All loop bounds are compile-time; A-fragments batched ahead of the MACs.
// Epilogue: each lane owns 8 consecutive couts -> one packed b128 store.
// ---------------------------------------------------------------------------
template <int CIN, int NMB>
__device__ __forceinline__ void conv_pass(
    const u16t* __restrict__ tile, const unsigned int* __restrict__ fragW,
    const float* __restrict__ bias, u16t* __restrict__ outp,
    int mb0, int MBtot, int orow, int ocol, int khalf, int lane,
    int CoutReal, int lrelu, long recBase) {
  constexpr int KC = CIN / 32;
  v8f acc[NMB] = {};
  for (int tap = 0; tap < 9; ++tap) {
    int ty_ = tap / 3, tx_ = tap - ty_ * 3;
    const u16t* bbase =
        &tile[((orow + ty_) * 18 + (ocol + tx_)) * CIN + khalf * 16];
    const unsigned int* abase0 =
        fragW + (((tap * KC) * MBtot + mb0) * 256 + lane * 8);
#pragma unroll
    for (int kc = 0; kc < KC; ++kc) {
      v16bf bf = *reinterpret_cast<const v16bf*>(bbase + kc * 32);
      const unsigned int* abase = abase0 + kc * MBtot * 256;
      v16bf af[NMB];
#pragma unroll
      for (int m = 0; m < NMB; ++m)
        af[m] = *reinterpret_cast<const v16bf*>(abase + m * 256);
#pragma unroll
      for (int m = 0; m < NMB; ++m)
        acc[m] = __builtin_amdgcn_wmma_f32_16x16x32_bf16(
            false, af[m], false, bf, (short)0, acc[m], false, false);
    }
  }
#pragma unroll
  for (int m = 0; m < NMB; ++m) {
    int cout0 = (mb0 + m) * 16 + khalf * 8;
    unsigned pk[4];
#pragma unroll
    for (int j = 0; j < 4; ++j) {
      float v0 = acc[m][2 * j]     + ((cout0 + 2 * j     < CoutReal) ? bias[cout0 + 2 * j]     : 0.f);
      float v1 = acc[m][2 * j + 1] + ((cout0 + 2 * j + 1 < CoutReal) ? bias[cout0 + 2 * j + 1] : 0.f);
      if (lrelu) {
        v0 = (v0 >= 0.f) ? v0 : 0.1f * v0;
        v1 = (v1 >= 0.f) ? v1 : 0.1f * v1;
      }
      pk[j] = pack2(v0, v1);
    }
    *reinterpret_cast<uint4*>(outp + recBase + cout0) =
        *reinterpret_cast<const uint4*>(pk);
  }
}

// ---------------------------------------------------------------------------
// 3x3 SAME conv as implicit GEMM on WMMA bf16, NHWC bf16 in/out.
// Block = 256 thr = 8 waves; output tile = 8 rows x 16 cols.
// LDS tile: (8+2)x(16+2) haloed pixel records x CIN bf16, [row][col][cin];
// staged as straight 16B chunks via global_load_async_to_lds_b128
// (OOB halo chunks zero-filled with ds_store_b128).
// ---------------------------------------------------------------------------
template <int CIN, int MBT>
__global__ void __launch_bounds__(256)
conv3x3_wmma(const u16t* __restrict__ in0, const u16t* __restrict__ in1,
             const unsigned int* __restrict__ fragW,
             const float* __restrict__ bias, u16t* __restrict__ outp,
             int H, int W, int CoutReal, int lrelu,
             int in1_center, int tT, int tCenter) {
  __shared__ __align__(16) u16t tile[10 * 18 * CIN];

  const int bt  = blockIdx.z;
  const int b   = bt / tT;
  const int bt1 = in1_center ? (b * tT + tCenter) : bt;
  const int tx = blockIdx.x, ty = blockIdx.y;
  const int tid  = threadIdx.x;
  const int lane = tid & 31, wave = tid >> 5;
  const long HW = (long)H * W;

  // ---- stage haloed NHWC tile to LDS (16B chunks) -------------------------
  const int row0 = ty * 8 - 1, col0 = tx * 16 - 1;
  constexpr int NPX = 10 * 18;
  constexpr int CPR = CIN / 8;               // 16B chunks per pixel record
  const unsigned ldsBase = (unsigned)(unsigned long long)(uintptr_t)tile;
  for (int ch = tid; ch < NPX * CPR; ch += 256) {
    int p   = ch / CPR;
    int sub = ch - p * CPR;
    int r = p / 18, q = p - r * 18;
    int gy = row0 + r, gx = col0 + q;
    if (gy >= 0 && gy < H && gx >= 0 && gx < W) {
      int half = sub >> 3;                   // which 64-ch source tensor
      int csub = sub & 7;
      const u16t* gsrc = (half == 0)
          ? in0 + ((long)bt  * HW + (long)gy * W + gx) * 64 + csub * 8
          : in1 + ((long)bt1 * HW + (long)gy * W + gx) * 64 + csub * 8;
#if USE_ASYNC_STAGING
      unsigned la = ldsBase + (unsigned)ch * 16u;
      unsigned long long ga = (unsigned long long)(uintptr_t)gsrc;
      asm volatile("global_load_async_to_lds_b128 %0, %1, off"
                   :: "v"(la), "v"(ga) : "memory");
#else
      *reinterpret_cast<uint4*>(tile + (size_t)ch * 8) =
          *reinterpret_cast<const uint4*>(gsrc);
#endif
    } else {
      uint4 z = {0u, 0u, 0u, 0u};
      *reinterpret_cast<uint4*>(tile + (size_t)ch * 8) = z;
    }
  }
#if USE_ASYNC_STAGING
  asm volatile("s_wait_asynccnt 0x0" ::: "memory");
#endif
  __syncthreads();

  const int orow  = wave;          // local output row (0..7)
  const int ocol  = lane & 15;     // local output col / WMMA N index
  const int khalf = lane >> 4;
  const int gy = ty * 8 + orow, gx = tx * 16 + ocol;
  const long recBase = ((long)bt * HW + (long)gy * W + gx) * (MBT * 16);

  if constexpr (MBT == 4) {
    conv_pass<CIN, 4>(tile, fragW, bias, outp, 0, MBT, orow, ocol, khalf,
                      lane, CoutReal, lrelu, recBase);
  } else {
    static_assert(MBT == 14, "unsupported MBT");
    conv_pass<CIN, 4>(tile, fragW, bias, outp, 0,  MBT, orow, ocol, khalf,
                      lane, CoutReal, lrelu, recBase);
    conv_pass<CIN, 4>(tile, fragW, bias, outp, 4,  MBT, orow, ocol, khalf,
                      lane, CoutReal, lrelu, recBase);
    conv_pass<CIN, 4>(tile, fragW, bias, outp, 8,  MBT, orow, ocol, khalf,
                      lane, CoutReal, lrelu, recBase);
    conv_pass<CIN, 2>(tile, fragW, bias, outp, 12, MBT, orow, ocol, khalf,
                      lane, CoutReal, lrelu, recBase);
  }
}

// ---------------------------------------------------------------------------
// Modulated deformable conv (DG=8, cg=8, K=9) as gather + WMMA GEMM.
// NHWC bf16 x / offsets.  A lane's two (g,k9) pairs are adjacent channels,
// so dy/dx/mask load as three aligned dwords; each bilinear corner is one
// b128 load (8 channels).  Output NHWC bf16, or f32 NCHW for the last stage.
// ---------------------------------------------------------------------------
__global__ void __launch_bounds__(256)
mdcn_wmma(const u16t* __restrict__ xin, const u16t* __restrict__ off,
          const unsigned int* __restrict__ fragW, const float* __restrict__ bias,
          void* __restrict__ outp, int H, int W, int lrelu, int out_nchw) {
  const int bt = blockIdx.z;
  const long HW = (long)H * W;
  const int lane = threadIdx.x & 31, wave = threadIdx.x >> 5;
  const int row = blockIdx.y * 8 + wave;
  const int col = blockIdx.x * 16 + (lane & 15);
  const int khalf = lane >> 4;
  const float rowf = (float)row, colf = (float)col;
  const long pix = (long)row * W + col;
  const u16t* xb = xin + (long)bt * HW * 64;
  const u16t* ob = off + ((long)bt * HW + pix) * 224;

  v8f acc[4] = {};
  union { v16bf v; unsigned d[8]; } bfr;

  for (int kc = 0; kc < 18; ++kc) {
    int pair0 = kc * 4 + khalf * 2;           // even: two adjacent pairs
    unsigned dyp = *reinterpret_cast<const unsigned*>(ob + pair0);
    unsigned dxp = *reinterpret_cast<const unsigned*>(ob + 72 + pair0);
    unsigned mvp = *reinterpret_cast<const unsigned*>(ob + 144 + pair0);
#pragma unroll
    for (int p2 = 0; p2 < 2; ++p2) {
      int pairIdx = pair0 + p2;               // 0..71  (= g*9 + k9)
      int g = pairIdx / 9, k9 = pairIdx - g * 9;
      float dy = p2 ? bfh(dyp) : bfl(dyp);
      float dx = p2 ? bfh(dxp) : bfl(dxp);
      float mv = p2 ? bfh(mvp) : bfl(mvp);
      float mask = 1.f / (1.f + __expf(-mv));
      float py = rowf + (float)(k9 / 3 - 1) + dy;
      float px = colf + (float)(k9 % 3 - 1) + dx;
      float fy0 = floorf(py), fx0 = floorf(px);
      int y0 = (int)fy0, x0 = (int)fx0;
      float wy = py - fy0, wx = px - fx0;
      int y1 = y0 + 1, x1 = x0 + 1;
      bool vy0 = (y0 >= 0) & (y0 < H), vy1 = (y1 >= 0) & (y1 < H);
      bool vx0 = (x0 >= 0) & (x0 < W), vx1 = (x1 >= 0) & (x1 < W);
      float w00 = (1.f - wy) * (1.f - wx) * ((vy0 & vx0) ? mask : 0.f);
      float w01 = (1.f - wy) * wx         * ((vy0 & vx1) ? mask : 0.f);
      float w10 = wy * (1.f - wx)         * ((vy1 & vx0) ? mask : 0.f);
      float w11 = wy * wx                 * ((vy1 & vx1) ? mask : 0.f);
      int cy0 = iclamp(y0, 0, H - 1), cy1 = iclamp(y1, 0, H - 1);
      int cx0 = iclamp(x0, 0, W - 1), cx1 = iclamp(x1, 0, W - 1);
      const u16t* xg = xb + g * 8;
      long p00 = ((long)cy0 * W + cx0) * 64, p01 = ((long)cy0 * W + cx1) * 64;
      long p10 = ((long)cy1 * W + cx0) * 64, p11 = ((long)cy1 * W + cx1) * 64;
      unsigned q00[4], q01[4], q10[4], q11[4];
      *reinterpret_cast<uint4*>(q00) = *reinterpret_cast<const uint4*>(xg + p00);
      *reinterpret_cast<uint4*>(q01) = *reinterpret_cast<const uint4*>(xg + p01);
      *reinterpret_cast<uint4*>(q10) = *reinterpret_cast<const uint4*>(xg + p10);
      *reinterpret_cast<uint4*>(q11) = *reinterpret_cast<const uint4*>(xg + p11);
#pragma unroll
      for (int j = 0; j < 4; ++j) {                 // channels 2j, 2j+1
        float vl = w00 * bfl(q00[j]) + w01 * bfl(q01[j]) +
                   w10 * bfl(q10[j]) + w11 * bfl(q11[j]);
        float vh = w00 * bfh(q00[j]) + w01 * bfh(q01[j]) +
                   w10 * bfh(q10[j]) + w11 * bfh(q11[j]);
        bfr.d[p2 * 4 + j] = pack2(vl, vh);
      }
    }
    const unsigned int* abase = fragW + ((long)kc * 4 * 256 + lane * 8);
    v16bf af[4];
#pragma unroll
    for (int m = 0; m < 4; ++m)
      af[m] = *reinterpret_cast<const v16bf*>(abase + m * 256);
#pragma unroll
    for (int m = 0; m < 4; ++m)
      acc[m] = __builtin_amdgcn_wmma_f32_16x16x32_bf16(
          false, af[m], false, bfr.v, (short)0, acc[m], false, false);
  }
  if (out_nchw) {
    float* o = (float*)outp;
#pragma unroll
    for (int m = 0; m < 4; ++m)
#pragma unroll
      for (int r = 0; r < 8; ++r) {
        int cout = m * 16 + khalf * 8 + r;
        float v = acc[m][r] + bias[cout];
        if (lrelu) v = (v >= 0.f) ? v : 0.1f * v;
        o[((long)bt * 64 + cout) * HW + pix] = v;
      }
  } else {
    u16t* o = (u16t*)outp;
    long recBase = ((long)bt * HW + pix) * 64;
#pragma unroll
    for (int m = 0; m < 4; ++m) {
      int cout0 = m * 16 + khalf * 8;
      unsigned pk[4];
#pragma unroll
      for (int j = 0; j < 4; ++j) {
        float v0 = acc[m][2 * j]     + bias[cout0 + 2 * j];
        float v1 = acc[m][2 * j + 1] + bias[cout0 + 2 * j + 1];
        if (lrelu) {
          v0 = (v0 >= 0.f) ? v0 : 0.1f * v0;
          v1 = (v1 >= 0.f) ? v1 : 0.1f * v1;
        }
        pk[j] = pack2(v0, v1);
      }
      *reinterpret_cast<uint4*>(o + recBase + cout0) =
          *reinterpret_cast<const uint4*>(pk);
    }
  }
}

// ---------------------------------------------------------------------------
// Layout/resample helpers (NHWC bf16).
// ---------------------------------------------------------------------------
__global__ void nchw2nhwc_k(const float* __restrict__ in, u16t* __restrict__ out,
                            int HW, long total) {
  long i = (long)blockIdx.x * 256 + threadIdx.x;
  if (i >= total) return;
  long c = i & 63;
  long btpix = i >> 6;
  long bt = btpix / HW, pix = btpix - bt * HW;
  out[i] = f32_to_bf16(in[(bt * 64 + c) * HW + pix]);
}

__global__ void avgpool2_nhwc_k(const u16t* __restrict__ in, u16t* __restrict__ out,
                                int H, int W, long total) {
  long i = (long)blockIdx.x * 256 + threadIdx.x;
  if (i >= total) return;
  int oW = W >> 1;
  long oHW = (long)(H >> 1) * oW;
  long c = i & 63;
  long btpix = i >> 6;
  long bt = btpix / oHW;
  long p = btpix - bt * oHW;
  int oy = (int)(p / oW), ox = (int)(p - (long)oy * oW);
  const u16t* s = in + (bt * (long)H * W + (long)(2 * oy) * W + 2 * ox) * 64 + c;
  long rs = (long)W * 64;
  float v = 0.25f * (bf2f(s[0]) + bf2f(s[64]) + bf2f(s[rs]) + bf2f(s[rs + 64]));
  out[i] = f32_to_bf16(v);
}

__global__ void up2_nhwc_k(const u16t* __restrict__ in, u16t* __restrict__ out,
                           int H, int W, float scale, long total) {
  long i = (long)blockIdx.x * 256 + threadIdx.x;
  if (i >= total) return;
  long HW = (long)H * W;
  long c = i & 63;
  long btpix = i >> 6;
  long bt = btpix / HW;
  long p = btpix - bt * HW;
  int y = (int)(p / W), x = (int)(p - (long)y * W);
  int iH = H >> 1, iW = W >> 1;
  float sy = 0.5f * (y + 0.5f) - 0.5f;
  float sx = 0.5f * (x + 0.5f) - 0.5f;
  float fy = floorf(sy), fx = floorf(sx);
  int y0 = (int)fy, x0 = (int)fx;
  float wy = sy - fy, wx = sx - fx;
  int y1 = iclamp(y0 + 1, 0, iH - 1), x1 = iclamp(x0 + 1, 0, iW - 1);
  y0 = iclamp(y0, 0, iH - 1); x0 = iclamp(x0, 0, iW - 1);
  const u16t* s = in + bt * (long)iH * iW * 64 + c;
  float v = (1.f - wy) * ((1.f - wx) * bf2f(s[((long)y0 * iW + x0) * 64]) +
                                 wx  * bf2f(s[((long)y0 * iW + x1) * 64]))
          +        wy  * ((1.f - wx) * bf2f(s[((long)y1 * iW + x0) * 64]) +
                                 wx  * bf2f(s[((long)y1 * iW + x1) * 64]));
  out[i] = f32_to_bf16(v * scale);
}

// ---------------------------------------------------------------------------
// Host orchestration.
// Input order assumes setup_inputs() insertion-order flattening:
//   x, then params {oa3,ob3,dcn3,oa2,ob2,dcn2,fb2,oa1,ob1,dcn1,fb1,ca,cb,cd},
//   conv_p -> (w,b); dcn_p -> (w,b,w_off,b_off).
// ---------------------------------------------------------------------------
enum {
  IN_X = 0,
  OA3_W, OA3_B, OB3_W, OB3_B,
  DCN3_W, DCN3_B, DCN3_WOFF, DCN3_BOFF,
  OA2_W, OA2_B, OB2_W, OB2_B,
  DCN2_W, DCN2_B, DCN2_WOFF, DCN2_BOFF,
  FB2_W, FB2_B,
  OA1_W, OA1_B, OB1_W, OB1_B,
  DCN1_W, DCN1_B, DCN1_WOFF, DCN1_BOFF,
  FB1_W, FB1_B,
  CA_W, CA_B, CB_W, CB_B,
  CD_W, CD_B, CD_WOFF, CD_BOFF
};

extern "C" void kernel_launch(void* const* d_in, const int* in_sizes, int n_in,
                              void* d_out, int out_size, void* d_ws, size_t ws_size,
                              hipStream_t stream) {
  (void)in_sizes; (void)n_in; (void)out_size; (void)ws_size;

  const float* X = (const float*)d_in[IN_X];
  auto F = [&](int i) { return (const float*)d_in[i]; };

  char* ws = (char*)d_ws;
  size_t off = 0;
  auto alloc = [&](size_t bytes) -> void* {
    void* p = ws + off;
    off = (off + bytes + 255) & ~(size_t)255;
    return p;
  };

  const int BT = 10, TT = 5, TC = 2;
  const long HW1 = 128 * 128, HW2 = 64 * 64, HW3 = 32 * 32;

  // ---- NHWC bf16 feature buffers ------------------------------------------
  auto ball = [&](long hw, int ch) {
    return (u16t*)alloc((size_t)BT * hw * ch * 2);
  };
  u16t* x1c = ball(HW1, 64);
  u16t* x2c = ball(HW2, 64);
  u16t* x3c = ball(HW3, 64);
  u16t* A3  = ball(HW3, 64);
  u16t* O3  = ball(HW3, 64);
  u16t* F3  = ball(HW3, 64);
  u16t* UO3 = ball(HW2, 64);
  u16t* UF3 = ball(HW2, 64);
  u16t* A2  = ball(HW2, 64);
  u16t* O2  = ball(HW2, 64);
  u16t* F2R = ball(HW2, 64);
  u16t* F2  = ball(HW2, 64);
  u16t* UO2 = ball(HW1, 64);
  u16t* UF2 = ball(HW1, 64);
  u16t* A1  = ball(HW1, 64);
  u16t* O1  = ball(HW1, 64);
  u16t* F1R = ball(HW1, 64);
  u16t* F1  = ball(HW1, 64);
  u16t* CAb = ball(HW1, 64);
  u16t* CBb = ball(HW1, 64);
  u16t* OFF = ball(HW1, 224);   // reused sequentially by all 4 DCN stages

  // ---- weight fragment buffers (uint = bf16 pair) -------------------------
  auto falloc = [&](int chunks, int mb) {
    return (unsigned int*)alloc((size_t)chunks * mb * 256 * 4);
  };
  unsigned int* f_oa3  = falloc(36, 4);   // Cin128
  unsigned int* f_ob3  = falloc(18, 4);   // Cin64
  unsigned int* f_off3 = falloc(18, 14);  // Cin64 -> 216(pad 224)
  unsigned int* f_d3   = falloc(18, 4);   // dcn
  unsigned int* f_oa2  = falloc(36, 4);
  unsigned int* f_ob2  = falloc(36, 4);
  unsigned int* f_off2 = falloc(18, 14);
  unsigned int* f_d2   = falloc(18, 4);
  unsigned int* f_fb2  = falloc(36, 4);
  unsigned int* f_oa1  = falloc(36, 4);
  unsigned int* f_ob1  = falloc(36, 4);
  unsigned int* f_off1 = falloc(18, 14);
  unsigned int* f_d1   = falloc(18, 4);
  unsigned int* f_fb1  = falloc(36, 4);
  unsigned int* f_ca   = falloc(36, 4);
  unsigned int* f_cb   = falloc(18, 4);
  unsigned int* f_offc = falloc(18, 14);
  unsigned int* f_dc   = falloc(18, 4);

  auto prep = [&](const float* w, unsigned int* frag, int Cin, int MB,
                  int CoutReal, int Ktot, int dcn) {
    int total = (Ktot / 32) * MB * 256;
    prepack_wmma<<<(total + 255) / 256, 256, 0, stream>>>(
        w, frag, Cin, MB, CoutReal, Ktot, dcn);
  };
  prep(F(OA3_W),    f_oa3,  128, 4, 64, 1152, 0);
  prep(F(OB3_W),    f_ob3,   64, 4, 64,  576, 0);
  prep(F(DCN3_WOFF),f_off3,  64, 14, 216, 576, 0);
  prep(F(DCN3_W),   f_d3,    64, 4, 64,  576, 1);
  prep(F(OA2_W),    f_oa2,  128, 4, 64, 1152, 0);
  prep(F(OB2_W),    f_ob2,  128, 4, 64, 1152, 0);
  prep(F(DCN2_WOFF),f_off2,  64, 14, 216, 576, 0);
  prep(F(DCN2_W),   f_d2,    64, 4, 64,  576, 1);
  prep(F(FB2_W),    f_fb2,  128, 4, 64, 1152, 0);
  prep(F(OA1_W),    f_oa1,  128, 4, 64, 1152, 0);
  prep(F(OB1_W),    f_ob1,  128, 4, 64, 1152, 0);
  prep(F(DCN1_WOFF),f_off1,  64, 14, 216, 576, 0);
  prep(F(DCN1_W),   f_d1,    64, 4, 64,  576, 1);
  prep(F(FB1_W),    f_fb1,  128, 4, 64, 1152, 0);
  prep(F(CA_W),     f_ca,   128, 4, 64, 1152, 0);
  prep(F(CB_W),     f_cb,    64, 4, 64,  576, 0);
  prep(F(CD_WOFF),  f_offc,  64, 14, 216, 576, 0);
  prep(F(CD_W),     f_dc,    64, 4, 64,  576, 1);

  // conv dispatch over template instantiations
  auto conv = [&](const u16t* in0, const u16t* in1, int in1c,
                  const unsigned int* frag, const float* bias, u16t* out,
                  int Hc, int Wc, int Cin, int MB, int CoutReal, int lrelu) {
    dim3 g(Wc / 16, Hc / 8, BT);
    const u16t* i1 = in1 ? in1 : in0;
    if (Cin == 128)
      conv3x3_wmma<128, 4><<<g, 256, 0, stream>>>(
          in0, i1, frag, bias, out, Hc, Wc, CoutReal, lrelu, in1c, TT, TC);
    else if (MB == 4)
      conv3x3_wmma<64, 4><<<g, 256, 0, stream>>>(
          in0, i1, frag, bias, out, Hc, Wc, CoutReal, lrelu, in1c, TT, TC);
    else
      conv3x3_wmma<64, 14><<<g, 256, 0, stream>>>(
          in0, i1, frag, bias, out, Hc, Wc, CoutReal, lrelu, in1c, TT, TC);
  };
  auto mdcn = [&](const u16t* xin, const u16t* offp, const unsigned int* frag,
                  const float* bias, void* out, int Hc, int Wc, int lrelu,
                  int nchw) {
    dim3 g(Wc / 16, Hc / 8, BT);
    mdcn_wmma<<<g, 256, 0, stream>>>(xin, offp, frag, bias, out, Hc, Wc,
                                     lrelu, nchw);
  };
  auto up2 = [&](const u16t* in, u16t* out, int Ho, int Wo, float scale) {
    long tot = (long)BT * Ho * Wo * 64;
    up2_nhwc_k<<<(int)((tot + 255) / 256), 256, 0, stream>>>(in, out, Ho, Wo,
                                                             scale, tot);
  };

  // ---- layout conversion + pyramids ---------------------------------------
  long t1 = (long)BT * HW1 * 64;
  nchw2nhwc_k<<<(int)((t1 + 255) / 256), 256, 0, stream>>>(X, x1c, (int)HW1, t1);
  long t2 = (long)BT * HW2 * 64;
  avgpool2_nhwc_k<<<(int)((t2 + 255) / 256), 256, 0, stream>>>(x1c, x2c, 128, 128, t2);
  long t3 = (long)BT * HW3 * 64;
  avgpool2_nhwc_k<<<(int)((t3 + 255) / 256), 256, 0, stream>>>(x2c, x3c, 64, 64, t3);

  // ---- level 3 ------------------------------------------------------------
  conv(x3c, x3c, 1, f_oa3, F(OA3_B), A3, 32, 32, 128, 4, 64, 1);        // oa3
  conv(A3, 0, 0, f_ob3, F(OB3_B), O3, 32, 32, 64, 4, 64, 1);           // ob3
  conv(O3, 0, 0, f_off3, F(DCN3_BOFF), OFF, 32, 32, 64, 14, 216, 0);   // offs
  mdcn(x3c, OFF, f_d3, F(DCN3_B), F3, 32, 32, 1, 0);                   // dcn3
  up2(O3, UO3, 64, 64, 2.f);
  up2(F3, UF3, 64, 64, 1.f);

  // ---- level 2 ------------------------------------------------------------
  conv(x2c, x2c, 1, f_oa2, F(OA2_B), A2, 64, 64, 128, 4, 64, 1);       // oa2
  conv(A2, UO3, 0, f_ob2, F(OB2_B), O2, 64, 64, 128, 4, 64, 1);        // ob2
  conv(O2, 0, 0, f_off2, F(DCN2_BOFF), OFF, 64, 64, 64, 14, 216, 0);   // offs
  mdcn(x2c, OFF, f_d2, F(DCN2_B), F2R, 64, 64, 0, 0);                  // dcn2
  conv(F2R, UF3, 0, f_fb2, F(FB2_B), F2, 64, 64, 128, 4, 64, 1);       // fb2
  up2(O2, UO2, 128, 128, 2.f);
  up2(F2, UF2, 128, 128, 1.f);

  // ---- level 1 ------------------------------------------------------------
  conv(x1c, x1c, 1, f_oa1, F(OA1_B), A1, 128, 128, 128, 4, 64, 1);     // oa1
  conv(A1, UO2, 0, f_ob1, F(OB1_B), O1, 128, 128, 128, 4, 64, 1);      // ob1
  conv(O1, 0, 0, f_off1, F(DCN1_BOFF), OFF, 128, 128, 64, 14, 216, 0); // offs
  mdcn(x1c, OFF, f_d1, F(DCN1_B), F1R, 128, 128, 0, 0);                // dcn1
  conv(F1R, UF2, 0, f_fb1, F(FB1_B), F1, 128, 128, 128, 4, 64, 0);     // fb1

  // ---- cascade ------------------------------------------------------------
  conv(F1, x1c, 1, f_ca, F(CA_B), CAb, 128, 128, 128, 4, 64, 1);       // ca
  conv(CAb, 0, 0, f_cb, F(CB_B), CBb, 128, 128, 64, 4, 64, 1);         // cb
  conv(CBb, 0, 0, f_offc, F(CD_BOFF), OFF, 128, 128, 64, 14, 216, 0);  // offs
  mdcn(F1, OFF, f_dc, F(CD_B), d_out, 128, 128, 1, 1);                 // cd
}
